// EntropyGuidedAttention_76184129896874
// MI455X (gfx1250) — matrix operand order repaired and checked by reading, not compiled, not run on hardware
//
#include <hip/hip_runtime.h>
#include <hip/hip_bf16.h>
#include <stdint.h>

// ---------------------------------------------------------------------------
// CDNA5 (gfx1250) bf16 WMMA types & helpers
// ---------------------------------------------------------------------------
typedef __attribute__((ext_vector_type(16))) __bf16        v16bf;
typedef __attribute__((ext_vector_type(8)))  float         v8f;
typedef __attribute__((ext_vector_type(8)))  unsigned int  v8u;
typedef unsigned short u16t;

// Pack two fp32 -> packed bf16 pair (lo = a, hi = b).
__device__ __forceinline__ unsigned int packbf2(float a, float b) {
#if __has_builtin(__builtin_amdgcn_cvt_pk_bf16_f32)
  auto r = __builtin_amdgcn_cvt_pk_bf16_f32(a, b);   // v_cvt_pk_bf16_f32
  return __builtin_bit_cast(unsigned int, r);
#else
  // round-half-up + byte-perm pack (3 VALU) -- confirmed path on ROCm 7.2
  unsigned int ua = __builtin_bit_cast(unsigned int, a) + 0x8000u;
  unsigned int ub = __builtin_bit_cast(unsigned int, b) + 0x8000u;
  return __builtin_amdgcn_perm(ub, ua, 0x07060302u); // {b.hi16, a.hi16}
#endif
}
__device__ __forceinline__ u16t f2bf(float f) {
  return (u16t)(packbf2(f, f) & 0xFFFFu);
}

__device__ __forceinline__ v8f wmma_bf16(v16bf a, v16bf b, v8f c) {
  // v_wmma_f32_16x16x32_bf16  D = A(16x32) * B(32x16) + C
  return __builtin_amdgcn_wmma_f32_16x16x32_bf16(
      /*neg_a=*/false, a, /*neg_b=*/false, b,
      /*c_mod=*/(short)0, c, /*reuse_a=*/false, /*reuse_b=*/false);
}

// A fragment (16x32 bf16, MxK) from a per-lane row pointer (row + h*8 folded):
// elems 0..7 -> K = k0+h*8.. ; elems 8..15 -> K = k0+16+h*8..
__device__ __forceinline__ v16bf frag_a_bf16(const u16t* __restrict__ rowp, int k0) {
  const u16t* p = rowp + k0;
  uint4 x = *(const uint4*)p;
  uint4 y = *(const uint4*)(p + 16);
  v8u u = {x.x, x.y, x.z, x.w, y.x, y.y, y.z, y.w};
  return __builtin_bit_cast(v16bf, u);
}
__device__ __forceinline__ v16bf frag_a_f32(const float* __restrict__ rowp, int k0) {
  const float* p = rowp + k0;
  float4 x0 = *(const float4*)p,        x1 = *(const float4*)(p + 4);
  float4 y0 = *(const float4*)(p + 16), y1 = *(const float4*)(p + 20);
  v8u u;
  u[0] = packbf2(x0.x, x0.y); u[1] = packbf2(x0.z, x0.w);
  u[2] = packbf2(x1.x, x1.y); u[3] = packbf2(x1.z, x1.w);
  u[4] = packbf2(y0.x, y0.y); u[5] = packbf2(y0.z, y0.w);
  u[6] = packbf2(y1.x, y1.y); u[7] = packbf2(y1.z, y1.w);
  return __builtin_bit_cast(v16bf, u);
}

// B fragment (32x16 bf16, KxN) from per-lane column pointer (col + h*8
// folded): elems 0..7 -> K = k0+h*8.. ; elems 8..15 -> K = k0+16+h*8..
__device__ __forceinline__ v16bf frag_b_bf16(const u16t* __restrict__ colp, int k0) {
  const u16t* p = colp + k0;
  uint4 x = *(const uint4*)p;
  uint4 y = *(const uint4*)(p + 16);
  v8u u = {x.x, x.y, x.z, x.w, y.x, y.y, y.z, y.w};
  return __builtin_bit_cast(v16bf, u);
}

// ---------------------------------------------------------------------------
// Kernel 0: elementwise fp32 -> bf16 (weights + activations pre-convert).
// Memory-bound: 2x b128 load -> 4 packs -> 1x b128 store per thread.
// ---------------------------------------------------------------------------
__global__ __launch_bounds__(256) void cvt_f32_bf16(const float* __restrict__ in,
                                                    u16t* __restrict__ out, int n8) {
  int i = blockIdx.x * 256 + threadIdx.x;
  if (i >= n8) return;
  const float4* p = (const float4*)in + 2 * (size_t)i;
  float4 x = p[0], y = p[1];
  uint4 o;
  o.x = packbf2(x.x, x.y); o.y = packbf2(x.z, x.w);
  o.z = packbf2(y.x, y.y); o.w = packbf2(y.z, y.w);
  ((uint4*)out)[i] = o;
}

// ---------------------------------------------------------------------------
// Kernel 1: out_bf16[M,768] = bf16( A[M,768] @ Wbf[768,768]^T + bias )
// A_IS_F32=true: convert A on the fly (low-ws fallback);
// A_IS_F32=false: pure bf16 operands (preferred; inner loop = loads + WMMA).
// transposed=1 writes out[col*M + row] (V path; 8 rows contiguous -> b128).
// Block = 256 thr = 8 waves (2x4), block tile 64x256, wave tile 32x64.
// ---------------------------------------------------------------------------
template <bool A_IS_F32>
__global__ __launch_bounds__(256) void gemm_bias_to_bf16(
    const void* __restrict__ Ap, const u16t* __restrict__ Wbf,
    const float* __restrict__ bias, u16t* __restrict__ out,
    int M, int transposed) {
  constexpr int D = 768;
  const int lane = threadIdx.x & 31, wid = threadIdx.x >> 5;
  const int wm = wid >> 2, wn = wid & 3;
  const int mBase = blockIdx.y * 64 + wm * 32;
  const int nBase = blockIdx.x * 256 + wn * 64;
  const int r16 = lane & 15, h = (lane >> 4) & 1;

  // loop-invariant per-lane row/col pointers
  const float* af0 = nullptr; const float* af1 = nullptr;
  const u16t*  ab0 = nullptr; const u16t*  ab1 = nullptr;
  if constexpr (A_IS_F32) {
    af0 = (const float*)Ap + (size_t)(mBase + r16) * D + h * 8;
    af1 = af0 + (size_t)16 * D;
  } else {
    ab0 = (const u16t*)Ap + (size_t)(mBase + r16) * D + h * 8;
    ab1 = ab0 + (size_t)16 * D;
  }
  const u16t* bcol[4];
#pragma unroll
  for (int t = 0; t < 4; ++t)
    bcol[t] = Wbf + (size_t)(nBase + t * 16 + r16) * D + h * 8;

  v8f z = 0;
  v8f c[2][4];
#pragma unroll
  for (int i = 0; i < 2; ++i)
#pragma unroll
    for (int t = 0; t < 4; ++t) c[i][t] = z;

#pragma unroll 8
  for (int k0 = 0; k0 < D; k0 += 32) {
    v16bf a0, a1;
    if constexpr (A_IS_F32) {
      a0 = frag_a_f32(af0, k0);
      a1 = frag_a_f32(af1, k0);
    } else {
      a0 = frag_a_bf16(ab0, k0);
      a1 = frag_a_bf16(ab1, k0);
    }
#pragma unroll
    for (int t = 0; t < 4; ++t) {
      v16bf bt = frag_b_bf16(bcol[t], k0);
      c[0][t] = wmma_bf16(a0, bt, c[0][t]);
      c[1][t] = wmma_bf16(a1, bt, c[1][t]);
    }
  }

#pragma unroll
  for (int i = 0; i < 2; ++i)
#pragma unroll
    for (int t = 0; t < 4; ++t) {
      const int col = nBase + t * 16 + r16;
      const float bv = bias[col];
      if (transposed) {
        uint4 o;
        o.x = packbf2(c[i][t][0] + bv, c[i][t][1] + bv);
        o.y = packbf2(c[i][t][2] + bv, c[i][t][3] + bv);
        o.z = packbf2(c[i][t][4] + bv, c[i][t][5] + bv);
        o.w = packbf2(c[i][t][6] + bv, c[i][t][7] + bv);
        *(uint4*)(out + (size_t)col * M + mBase + i * 16 + h * 8) = o;
      } else {
#pragma unroll
        for (int r = 0; r < 8; ++r) {
          const int row = mBase + i * 16 + r + h * 8;
          out[(size_t)row * D + col] = f2bf(c[i][t][r] + bv);
        }
      }
    }
}

// ---------------------------------------------------------------------------
// Kernel 2: two-pass streamed softmax producing P (bf16) and 1/rowsum.
// Wave owns 16 query rows; K streamed in 64-key chunks (4 WMMA acc tiles).
// Q/K/P all sit in the 192MB L2, so the pass-2 recompute is cheap.
// ---------------------------------------------------------------------------
__device__ __forceinline__ void qk_tile(const u16t* __restrict__ qrow,
                                        const u16t* __restrict__ Kb,
                                        int kc, int lane, v8f c[4]) {
  constexpr int D = 768;
  const int r16 = lane & 15, h = (lane >> 4) & 1;
  const u16t* kcol[4];
#pragma unroll
  for (int t = 0; t < 4; ++t)
    kcol[t] = Kb + (size_t)(kc + t * 16 + r16) * D + h * 8;

  v8f z = 0;
#pragma unroll
  for (int t = 0; t < 4; ++t) c[t] = z;
#pragma unroll 8
  for (int k0 = 0; k0 < D; k0 += 32) {
    v16bf a = frag_a_bf16(qrow, k0);
#pragma unroll
    for (int t = 0; t < 4; ++t) {
      v16bf bt = frag_b_bf16(kcol[t], k0);
      c[t] = wmma_bf16(a, bt, c[t]);
    }
  }
}

__global__ __launch_bounds__(256) void attn_softmax_p(
    const u16t* __restrict__ Q, const u16t* __restrict__ K,
    const float* __restrict__ ew, u16t* __restrict__ P,
    float* __restrict__ linv) {
  constexpr int D = 768, NT = 1024, NV = 4096;
  const float scale = 0.03608439182435161f;  // 1/sqrt(768)
  const int lane = threadIdx.x & 31, wid = threadIdx.x >> 5;
  const int b = blockIdx.y;
  const int mBase = blockIdx.x * 128 + wid * 16;
  const u16t* Kb = K + (size_t)b * NT * D;
  const float* ewb = ew + b * NT;
  const int col16 = lane & 15, hs = (lane >> 4) & 1;
  const u16t* qrow =
      Q + ((size_t)b * NV + mBase + col16) * D + hs * 8;  // per-lane Q row ptr

  float vmax[8];
#pragma unroll
  for (int r = 0; r < 8; ++r) vmax[r] = -1e30f;

  // ---- pass 1: row max of scale*w[k]*(q.k) ----
  for (int kc = 0; kc < NT; kc += 64) {
    v8f c[4];
    qk_tile(qrow, Kb, kc, lane, c);
#pragma unroll
    for (int t = 0; t < 4; ++t) {
      const float w = scale * ewb[kc + t * 16 + col16];
#pragma unroll
      for (int r = 0; r < 8; ++r) vmax[r] = fmaxf(vmax[r], c[t][r] * w);
    }
  }
#pragma unroll
  for (int r = 0; r < 8; ++r) {
#pragma unroll
    for (int off = 1; off < 16; off <<= 1)
      vmax[r] = fmaxf(vmax[r], __shfl_xor(vmax[r], off, 16));
  }

  // ---- pass 2: P = exp(s - max), accumulate row sums ----
  float lsum[8];
#pragma unroll
  for (int r = 0; r < 8; ++r) lsum[r] = 0.0f;

  for (int kc = 0; kc < NT; kc += 64) {
    v8f c[4];
    qk_tile(qrow, Kb, kc, lane, c);
#pragma unroll
    for (int t = 0; t < 4; ++t) {
      const int col = kc + t * 16 + col16;
      const float w = scale * ewb[col];
#pragma unroll
      for (int r = 0; r < 8; ++r) {
        const float p = __expf(c[t][r] * w - vmax[r]);
        lsum[r] += p;
        const int row = mBase + r + hs * 8;
        P[((size_t)b * NV + row) * NT + col] = f2bf(p);
      }
    }
  }
#pragma unroll
  for (int r = 0; r < 8; ++r) {
#pragma unroll
    for (int off = 1; off < 16; off <<= 1)
      lsum[r] += __shfl_xor(lsum[r], off, 16);
  }
  if (col16 == 0) {
#pragma unroll
    for (int r = 0; r < 8; ++r) {
      const int row = mBase + r + hs * 8;
      linv[b * NV + row] = 1.0f / lsum[r];
    }
  }
}

// ---------------------------------------------------------------------------
// Kernel 3: O[b] = (P[b] @ V[b]) * (1/l[row]).  V stored transposed as
// Vt[768, 8192] (batch b = columns b*1024..), so B-frag loads are contiguous.
// Block = 8 waves (2x4), block tile 64x256, wave tile 32x64.
// ---------------------------------------------------------------------------
__global__ __launch_bounds__(256) void pv_gemm(
    const u16t* __restrict__ P, const u16t* __restrict__ Vt,
    const float* __restrict__ linv, float* __restrict__ out) {
  constexpr int D = 768, NT = 1024, NV = 4096, LDT = 8192;
  const int lane = threadIdx.x & 31, wid = threadIdx.x >> 5;
  const int wm = wid >> 2, wn = wid & 3;
  const int b = blockIdx.z;
  const int mBase = blockIdx.y * 64 + wm * 32;
  const int nBase = blockIdx.x * 256 + wn * 64;
  const int r16 = lane & 15, h = (lane >> 4) & 1;

  const u16t* prow0 = P + ((size_t)b * NV + mBase + r16) * NT + h * 8;
  const u16t* prow1 = prow0 + (size_t)16 * NT;
  const u16t* vcol[4];
#pragma unroll
  for (int t = 0; t < 4; ++t)
    vcol[t] = Vt + (size_t)(nBase + t * 16 + r16) * LDT + b * NT + h * 8;

  v8f z = 0;
  v8f c[2][4];
#pragma unroll
  for (int i = 0; i < 2; ++i)
#pragma unroll
    for (int t = 0; t < 4; ++t) c[i][t] = z;

#pragma unroll 8
  for (int k0 = 0; k0 < NT; k0 += 32) {
    v16bf a0 = frag_a_bf16(prow0, k0);
    v16bf a1 = frag_a_bf16(prow1, k0);
#pragma unroll
    for (int t = 0; t < 4; ++t) {
      v16bf bt = frag_b_bf16(vcol[t], k0);
      c[0][t] = wmma_bf16(a0, bt, c[0][t]);
      c[1][t] = wmma_bf16(a1, bt, c[1][t]);
    }
  }

#pragma unroll
  for (int i = 0; i < 2; ++i) {
#pragma unroll
    for (int r = 0; r < 8; ++r) {
      const int row = mBase + i * 16 + r + h * 8;
      const float li = linv[b * NV + row];
#pragma unroll
      for (int t = 0; t < 4; ++t) {
        const int col = nBase + t * 16 + r16;
        out[((size_t)b * NV + row) * D + col] = c[i][t][r] * li;
      }
    }
  }
}

// ---------------------------------------------------------------------------
// Launch.  Workspace layout (bytes):
//   base:  Q bf16 50.3MB | K bf16 12.6MB | Vt bf16 12.6MB | P bf16 67.1MB
//          linv f32 0.13MB | Wq/Wk/Wv bf16 3.5MB            = ~146 MB
//   opt:   visual bf16 50.3MB | textual bf16 12.6MB         = ~209 MB total
// If ws_size covers the optional segment we pre-convert activations and run
// the all-bf16 projection GEMM; else fall back to fp32-A on-the-fly variant.
// Both fit in the 192MB L2 for the hot phase (P + K/V/Q slices).
// ---------------------------------------------------------------------------
extern "C" void kernel_launch(void* const* d_in, const int* in_sizes, int n_in,
                              void* d_out, int out_size, void* d_ws, size_t ws_size,
                              hipStream_t stream) {
  const float* visual  = (const float*)d_in[0];
  const float* textual = (const float*)d_in[1];
  const float* ew      = (const float*)d_in[2];
  const float* Wq      = (const float*)d_in[3];
  const float* bq      = (const float*)d_in[4];
  const float* Wk      = (const float*)d_in[5];
  const float* bk      = (const float*)d_in[6];
  const float* Wv      = (const float*)d_in[7];
  const float* bv      = (const float*)d_in[8];
  float* out = (float*)d_out;

  constexpr size_t B = 8, NV = 4096, NT = 1024, D = 768;
  u16t*  Qw   = (u16t*)d_ws;
  u16t*  Kw   = Qw  + B * NV * D;
  u16t*  Vtw  = Kw  + B * NT * D;
  u16t*  Pw   = Vtw + D * (B * NT);
  float* linv = (float*)(Pw + B * NV * NT);
  u16t*  Wqb  = (u16t*)(linv + B * NV);
  u16t*  Wkb  = Wqb + D * D;
  u16t*  Wvb  = Wkb + D * D;
  u16t*  Visb = Wvb + D * D;            // optional segment
  u16t*  Txtb = Visb + B * NV * D;
  const size_t need_full = (size_t)((char*)(Txtb + B * NT * D) - (char*)d_ws);
  const bool preconv = ws_size >= need_full;

  dim3 blk(256);
  const int wN8 = (int)(D * D / 8);          // 73728 packs of 8
  const int wGrid = (wN8 + 255) / 256;       // 288 blocks

  // fp32 -> bf16 weight pre-convert (one pass each, tiny)
  cvt_f32_bf16<<<dim3(wGrid), blk, 0, stream>>>(Wq, Wqb, wN8);
  cvt_f32_bf16<<<dim3(wGrid), blk, 0, stream>>>(Wk, Wkb, wN8);
  cvt_f32_bf16<<<dim3(wGrid), blk, 0, stream>>>(Wv, Wvb, wN8);

  if (preconv) {
    // activations -> bf16 once (memory-bound), then all-bf16 GEMMs
    const int vN8 = (int)(B * NV * D / 8), tN8 = (int)(B * NT * D / 8);
    cvt_f32_bf16<<<dim3((vN8 + 255) / 256), blk, 0, stream>>>(visual, Visb, vN8);
    cvt_f32_bf16<<<dim3((tN8 + 255) / 256), blk, 0, stream>>>(textual, Txtb, tN8);
    gemm_bias_to_bf16<false><<<dim3(3, (B * NV) / 64), blk, 0, stream>>>(
        Visb, Wqb, bq, Qw, (int)(B * NV), 0);
    gemm_bias_to_bf16<false><<<dim3(3, (B * NT) / 64), blk, 0, stream>>>(
        Txtb, Wkb, bk, Kw, (int)(B * NT), 0);
    gemm_bias_to_bf16<false><<<dim3(3, (B * NT) / 64), blk, 0, stream>>>(
        Txtb, Wvb, bv, Vtw, (int)(B * NT), 1);
  } else {
    // low-workspace fallback: convert A inside the GEMM
    gemm_bias_to_bf16<true><<<dim3(3, (B * NV) / 64), blk, 0, stream>>>(
        visual, Wqb, bq, Qw, (int)(B * NV), 0);
    gemm_bias_to_bf16<true><<<dim3(3, (B * NT) / 64), blk, 0, stream>>>(
        textual, Wkb, bk, Kw, (int)(B * NT), 0);
    gemm_bias_to_bf16<true><<<dim3(3, (B * NT) / 64), blk, 0, stream>>>(
        textual, Wvb, bv, Vtw, (int)(B * NT), 1);
  }

  // softmax(scale * w ⊙ QK^T) -> P bf16 + 1/rowsum
  attn_softmax_p<<<dim3(NV / 128, B), blk, 0, stream>>>(Qw, Kw, ew, Pw, linv);
  // O = (P @ V) * (1/l)
  pv_gemm<<<dim3(3, NV / 64, B), blk, 0, stream>>>(Pw, Vtw, linv, out);
}
